// HierarchicalVAE_83717502534357
// MI455X (gfx1250) — matrix operand order repaired and checked by reading, not compiled, not run on hardware
//
#include <hip/hip_runtime.h>
#include <hip/hip_bf16.h>

// ---------------------------------------------------------------------------
// Hierarchical VAE forward for MI455X (gfx1250, wave32, WMMA).
// Heavy GEMMs use v_wmma_f32_16x16x32_bf16 (f32 accumulate, bf16 operands
// converted on the fly), with double-buffered LDS staging so global-load
// latency overlaps WMMA compute.  Everything f32 in global memory.
// ---------------------------------------------------------------------------

typedef __attribute__((ext_vector_type(16))) __bf16 bf16x16;
typedef __attribute__((ext_vector_type(8)))  float  f32x8;

#define CDIV(a, b) (((a) + (b) - 1) / (b))

union BfPair {
    __bf16   h[2];
    unsigned u;
};

// ---------------- Generic tiled WMMA GEMM:  C = act(A @ B + bias) ----------
// A: MxK f32 row-major, B: KxN f32 row-major, C: MxN f32 row-major.
// Block tile 64x64, 4 waves of 32 lanes, each wave computes a 16x64 strip.
// K staged in chunks of 32 as bf16 in ping-pong LDS buffers:
//   iter c: global-load chunk c+1 into registers, compute chunk c from LDS,
//           convert+store chunk c+1, one barrier.
// ACT: 0 = identity, 1 = relu, 2 = sigmoid.

#define BM 64
#define BN 64
#define BK 32

template <int ACT>
__global__ __launch_bounds__(128) void gemm_bf16_wmma(
    const float* __restrict__ A, const float* __restrict__ B,
    const float* __restrict__ bias, float* __restrict__ C,
    int M, int N, int K)
{
    __shared__ __bf16 As[2][BM * BK];   // [buf][row][k]
    __shared__ __bf16 Bs[2][BN * BK];   // [buf][col][k]  (transposed)

    const int tid  = threadIdx.x;
    const int lane = tid & 31;
    const int wave = tid >> 5;
    const int m0   = blockIdx.y * BM;
    const int n0   = blockIdx.x * BN;
    const bool mn_full = (m0 + BM <= M) && (n0 + BN <= N);
    const int nc = CDIV(K, BK);

    f32x8 acc[4] = {};                  // four 16x16 C tiles per wave

    // staging registers for one K-chunk (distinct regs -> batched loads)
    float a0[8], a1[8], b0[8], b1[8];

    // --- phase helpers ------------------------------------------------------
    // A chunk: 64x32 elements as 1024 k-adjacent pairs; 8 pairs per thread.
    // B chunk: 32x64 elements as 1024 k-adjacent pairs of the transposed tile.
    auto preload = [&](int kb) {
        if (mn_full && (kb + BK <= K)) {
            #pragma unroll
            for (int i = 0; i < 8; ++i) {
                int idx2 = i * 128 + tid;
                int r = idx2 >> 4, k0 = (idx2 & 15) << 1;
                const float* ap = A + (long)(m0 + r) * K + (kb + k0);
                a0[i] = ap[0];
                a1[i] = ap[1];
            }
            #pragma unroll
            for (int i = 0; i < 8; ++i) {
                int idx2 = i * 128 + tid;
                int c = idx2 & 63, k0 = (idx2 >> 6) << 1;
                const float* bp = B + (long)(kb + k0) * N + (n0 + c);
                b0[i] = bp[0];
                b1[i] = bp[N];
            }
        } else {
            #pragma unroll
            for (int i = 0; i < 8; ++i) {
                int idx2 = i * 128 + tid;
                int r = idx2 >> 4, k0 = (idx2 & 15) << 1;
                int gr = m0 + r;
                const float* ap = A + (long)gr * K + (kb + k0);
                a0[i] = (gr < M && kb + k0     < K) ? ap[0] : 0.0f;
                a1[i] = (gr < M && kb + k0 + 1 < K) ? ap[1] : 0.0f;
            }
            #pragma unroll
            for (int i = 0; i < 8; ++i) {
                int idx2 = i * 128 + tid;
                int c = idx2 & 63, k0 = (idx2 >> 6) << 1;
                int gc = n0 + c;
                const float* bp = B + (long)(kb + k0) * N + gc;
                b0[i] = (gc < N && kb + k0     < K) ? bp[0] : 0.0f;
                b1[i] = (gc < N && kb + k0 + 1 < K) ? bp[N] : 0.0f;
            }
        }
    };

    auto store_lds = [&](int buf) {
        #pragma unroll
        for (int i = 0; i < 8; ++i) {
            int idx2 = i * 128 + tid;
            int r = idx2 >> 4, k0 = (idx2 & 15) << 1;
            BfPair p;
            p.h[0] = (__bf16)a0[i];
            p.h[1] = (__bf16)a1[i];
            *reinterpret_cast<unsigned*>(&As[buf][r * BK + k0]) = p.u;
        }
        #pragma unroll
        for (int i = 0; i < 8; ++i) {
            int idx2 = i * 128 + tid;
            int c = idx2 & 63, k0 = (idx2 >> 6) << 1;
            BfPair p;
            p.h[0] = (__bf16)b0[i];
            p.h[1] = (__bf16)b1[i];
            *reinterpret_cast<unsigned*>(&Bs[buf][c * BK + k0]) = p.u;
        }
    };

    auto compute = [&](int buf) {
        // A fragment: 16x32, contiguous k order (same as B)
        // lane L: M = L%16 ; VGPR v holds K = (L/16)*16 + 2v, +1
        bf16x16 afrag;
        {
            const __bf16* Ar = &As[buf][(wave * 16 + (lane & 15)) * BK];
            const int hb = (lane >> 4) * 16;
            #pragma unroll
            for (int v = 0; v < 8; ++v) {
                int k0 = hb + (v << 1);
                afrag[2 * v]     = Ar[k0];
                afrag[2 * v + 1] = Ar[k0 + 1];
            }
        }
        // All four B fragments into distinct registers first so the ds_loads
        // overlap and the four WMMAs issue back-to-back on one dscnt wait.
        // lane L: N = L%16 ; VGPR v holds K = (L/16)*16 + 2v, +1
        bf16x16 bfrag[4];
        #pragma unroll
        for (int nt = 0; nt < 4; ++nt) {
            const __bf16* Br = &Bs[buf][(nt * 16 + (lane & 15)) * BK];
            const int kof = (lane >> 4) * 16;
            #pragma unroll
            for (int v = 0; v < 8; ++v) {
                bfrag[nt][2 * v]     = Br[kof + 2 * v];
                bfrag[nt][2 * v + 1] = Br[kof + 2 * v + 1];
            }
        }
        #pragma unroll
        for (int nt = 0; nt < 4; ++nt) {
            acc[nt] = __builtin_amdgcn_wmma_f32_16x16x32_bf16(
                false, afrag, false, bfrag[nt], (short)0, acc[nt], false, false);
        }
    };

    // --- software-pipelined main loop --------------------------------------
    preload(0);
    store_lds(0);
    __syncthreads();
    for (int c = 0; c < nc; ++c) {
        if (c + 1 < nc) preload((c + 1) * BK);   // loads in flight during compute
        compute(c & 1);
        if (c + 1 < nc) store_lds((c + 1) & 1);  // other buffer: no race with readers
        __syncthreads();
    }

    // --- epilogue: C layout VGPR r -> M = r + (lane/16)*8, N = lane%16 ---
    const int rbase = m0 + wave * 16 + ((lane >> 4) << 3);
    const int cbase = n0 + (lane & 15);
    if (mn_full) {
        #pragma unroll
        for (int nt = 0; nt < 4; ++nt) {
            int col = cbase + nt * 16;
            float bv = bias[col];
            #pragma unroll
            for (int r = 0; r < 8; ++r) {
                float v = acc[nt][r] + bv;
                if (ACT == 1) v = v > 0.0f ? v : 0.0f;
                else if (ACT == 2) v = 1.0f / (1.0f + __expf(-v));
                C[(long)(rbase + r) * N + col] = v;
            }
        }
    } else {
        #pragma unroll
        for (int nt = 0; nt < 4; ++nt) {
            int col = cbase + nt * 16;
            if (col >= N) continue;
            float bv = bias[col];
            #pragma unroll
            for (int r = 0; r < 8; ++r) {
                int row = rbase + r;
                if (row < M) {
                    float v = acc[nt][r] + bv;
                    if (ACT == 1) v = v > 0.0f ? v : 0.0f;
                    else if (ACT == 2) v = 1.0f / (1.0f + __expf(-v));
                    C[(long)row * N + col] = v;
                }
            }
        }
    }
}

// ---------------- Patch extraction --------------------------------------
// out[row, e] = x_elem(e) + EPS * sum(channel window), row = b*441 + patch.
__global__ __launch_bounds__(96) void patch_kernel(const float* __restrict__ x,
                                                   float* __restrict__ pflat)
{
    __shared__ float sm[80];
    const int row = blockIdx.x;           // 0 .. 112895
    const int b = row / 441, p = row % 441;
    const int pr = p / 21, pc = p % 21;
    const int oy = pr * 3 - 1, ox = pc * 3 - 1;
    const int e = threadIdx.x;
    float val = 0.0f;
    if (e < 75) {
        int c = e / 25, j = e % 25, dy = j / 5, dx = j % 5;
        int iy = oy + dy, ix = ox + dx;
        if (iy >= 0 && iy < 63 && ix >= 0 && ix < 63)
            val = x[((b * 3 + c) * 63 + iy) * 63 + ix];
        sm[e] = val;
    }
    __syncthreads();
    if (e < 75) {
        int c = e / 25;
        float s = 0.0f;
        #pragma unroll
        for (int j = 0; j < 25; ++j) s += sm[c * 25 + j];
        pflat[(long)row * 75 + e] = val + 1e-6f * s;
    }
}

// ---------------- Reparameterizations / gathers / assembly ---------------
__global__ void little_reparam_kernel(const float* __restrict__ stats,
                                      const float* __restrict__ noise,
                                      float* __restrict__ z_out, int n)
{
    int idx = blockIdx.x * blockDim.x + threadIdx.x;
    if (idx >= n) return;
    int r = idx >> 4, j = idx & 15;
    float m  = stats[r * 32 + j];
    float lv = stats[r * 32 + 16 + j];
    z_out[idx] = noise[idx] * __expf(0.5f * lv) + m;
}

__global__ void enc_reparam_kernel(const float* __restrict__ stats,
                                   const float* __restrict__ noise,
                                   float* __restrict__ mean_out,
                                   float* __restrict__ logvar_out,
                                   float* __restrict__ z_out, int n)
{
    int idx = blockIdx.x * blockDim.x + threadIdx.x;
    if (idx >= n) return;
    int b = idx >> 6, j = idx & 63;
    float m  = stats[b * 128 + j];
    float lv = stats[b * 128 + 64 + j];
    mean_out[idx]   = m;
    logvar_out[idx] = lv;
    z_out[idx]      = noise[idx] * __expf(0.5f * lv) + m;
}

__global__ void gather_zt_kernel(const float* __restrict__ latent,
                                 float* __restrict__ zt, int n)
{
    int idx = blockIdx.x * blockDim.x + threadIdx.x;
    if (idx >= n) return;
    int r = idx >> 4, j = idx & 15;
    int b = r / 169, t = r % 169;
    zt[idx] = latent[((long)(b * 441 + t)) * 16 + j];
}

__global__ void img_kernel(const float* __restrict__ tiles,
                           float* __restrict__ img, int n)
{
    int idx = blockIdx.x * blockDim.x + threadIdx.x;
    if (idx >= n) return;
    // idx enumerates (b, c, y, x) with y, x in [0, 64)
    int xq = idx & 63;
    int yq = (idx >> 6) & 63;
    int bc = idx >> 12;
    int b  = bc / 3, c = bc % 3;
    int ty = yq / 5, py = yq % 5;
    int tx = xq / 5, px = xq % 5;
    long r = (long)b * 169 + ty * 13 + tx;
    img[idx] = tiles[r * 75 + c * 25 + py * 5 + px];
}

// ---------------------------------------------------------------------------
extern "C" void kernel_launch(void* const* d_in, const int* in_sizes, int n_in,
                              void* d_out, int out_size, void* d_ws, size_t ws_size,
                              hipStream_t stream)
{
    const float* x      = (const float*)d_in[0];
    const float* nl     = (const float*)d_in[1];
    const float* nz     = (const float*)d_in[2];
    const float* le_w1  = (const float*)d_in[3];
    const float* le_b1  = (const float*)d_in[4];
    const float* le_w2  = (const float*)d_in[5];
    const float* le_b2  = (const float*)d_in[6];
    const float* ld_w1  = (const float*)d_in[7];
    const float* ld_b1  = (const float*)d_in[8];
    const float* ld_w2  = (const float*)d_in[9];
    const float* ld_b2  = (const float*)d_in[10];
    const float* enc_w1 = (const float*)d_in[11];
    const float* enc_b1 = (const float*)d_in[12];
    const float* enc_w2 = (const float*)d_in[13];
    const float* enc_b2 = (const float*)d_in[14];
    const float* dec_w1 = (const float*)d_in[15];
    const float* dec_b1 = (const float*)d_in[16];
    const float* dec_w2 = (const float*)d_in[17];
    const float* dec_b2 = (const float*)d_in[18];
    const float* dec_w3 = (const float*)d_in[19];
    const float* dec_b3 = (const float*)d_in[20];

    // outputs (flat, concatenated in return order)
    float* out      = (float*)d_out;
    float* latent   = out;                 // 256*7056 = 1,806,336
    float* mean_o   = out + 1806336;       // 16,384
    float* lv_o     = out + 1822720;       // 16,384
    float* nli      = out + 1839104;       // 1,806,336 (next_layer_input == z_l)
    float* img      = out + 3645440;       // 3,145,728

    // workspace layout (floats), with lifetime-based reuse; peak ~92 MB
    float* ws     = (float*)d_ws;
    float* pflat  = ws + 0;               // 8,467,200   (dead after G1)
    float* h1     = ws + 8500000;         // 14,450,688  (dead after G2)
    float* statsl = ws + 0;               // 3,612,672   (reuse pflat; dead after reparam)
    float* henc   = ws + 0;               // 1,048,576   (reuse; after little reparam)
    float* statse = ws + 1100000;         // 32,768
    float* zbig   = ws + 1150000;         // 16,384
    float* dh1    = ws + 1170000;         // 8,192
    float* dh2    = ws + 1200000;         // 1,048,576
    float* zt     = ws + 2300000;         // 692,224
    float* ldh    = ws + 3000000;         // 5,537,792   (reuses dead h1 region)
    float* tiles  = ws + 8600000;         // 3,244,800

    const int BATCH = 256;
    const int ROWS_L = BATCH * 441;       // 112,896
    const int ROWS_D = BATCH * 169;       // 43,264

    // 1. patch extraction -> pflat (112896 x 75)
    patch_kernel<<<ROWS_L, 96, 0, stream>>>(x, pflat);

    // 2. little encoder: relu(pflat @ le_w1 + b1) -> h1 ; h1 @ le_w2 + b2 -> statsl
    gemm_bf16_wmma<1><<<dim3(CDIV(128, BN), CDIV(ROWS_L, BM)), 128, 0, stream>>>(
        pflat, le_w1, le_b1, h1, ROWS_L, 128, 75);
    gemm_bf16_wmma<0><<<dim3(CDIV(32, BN), CDIV(ROWS_L, BM)), 128, 0, stream>>>(
        h1, le_w2, le_b2, statsl, ROWS_L, 32, 128);

    // 3. little reparam -> next_layer_input output (== big encoder A matrix)
    {
        int n = ROWS_L * 16;
        little_reparam_kernel<<<CDIV(n, 256), 256, 0, stream>>>(statsl, nl, nli, n);
    }

    // 4. big encoder: relu(z_l @ enc_w1 + b1) -> henc ; relu(henc @ enc_w2 + b2) -> statse
    gemm_bf16_wmma<1><<<dim3(CDIV(4096, BN), CDIV(BATCH, BM)), 128, 0, stream>>>(
        nli, enc_w1, enc_b1, henc, BATCH, 4096, 7056);
    gemm_bf16_wmma<1><<<dim3(CDIV(128, BN), CDIV(BATCH, BM)), 128, 0, stream>>>(
        henc, enc_w2, enc_b2, statse, BATCH, 128, 4096);

    // 5. big reparam -> mean, logvar outputs, z
    {
        int n = BATCH * 64;
        enc_reparam_kernel<<<CDIV(n, 256), 256, 0, stream>>>(statse, nz, mean_o, lv_o, zbig, n);
    }

    // 6. decoder: relu(z @ dec_w1) -> dh1 ; relu(dh1 @ dec_w2) -> dh2 ;
    //    relu(dh2 @ dec_w3) -> latent (output)
    gemm_bf16_wmma<1><<<dim3(CDIV(32, BN), CDIV(BATCH, BM)), 128, 0, stream>>>(
        zbig, dec_w1, dec_b1, dh1, BATCH, 32, 64);
    gemm_bf16_wmma<1><<<dim3(CDIV(4096, BN), CDIV(BATCH, BM)), 128, 0, stream>>>(
        dh1, dec_w2, dec_b2, dh2, BATCH, 4096, 32);
    gemm_bf16_wmma<1><<<dim3(CDIV(7056, BN), CDIV(BATCH, BM)), 128, 0, stream>>>(
        dh2, dec_w3, dec_b3, latent, BATCH, 7056, 4096);

    // 7. gather decode tiles' latents (first 169 patches per batch)
    {
        int n = ROWS_D * 16;
        gather_zt_kernel<<<CDIV(n, 256), 256, 0, stream>>>(latent, zt, n);
    }

    // 8. little decoder: relu(zt @ ld_w1 + b1) -> ldh ; sigmoid(ldh @ ld_w2 + b2) -> tiles
    gemm_bf16_wmma<1><<<dim3(CDIV(128, BN), CDIV(ROWS_D, BM)), 128, 0, stream>>>(
        zt, ld_w1, ld_b1, ldh, ROWS_D, 128, 16);
    gemm_bf16_wmma<2><<<dim3(CDIV(75, BN), CDIV(ROWS_D, BM)), 128, 0, stream>>>(
        ldh, ld_w2, ld_b2, tiles, ROWS_D, 75, 128);

    // 9. assemble image: (B,13,13,3,5,5) -> (B,3,65,65) cropped to 64x64
    {
        int n = BATCH * 3 * 64 * 64;
        img_kernel<<<CDIV(n, 256), 256, 0, stream>>>(tiles, img, n);
    }
}
